// ScaledDotProductAttention_58411555226190
// MI455X (gfx1250) — matrix operand order repaired and checked by reading, not compiled, not run on hardware
//
#include <hip/hip_runtime.h>
#include <hip/hip_bf16.h>

// ---------------------------------------------------------------------------
// Fused scaled-dot-product attention for gfx1250 (MI455X), wave32 + WMMA.
//   B=2, H=16, T=2048, C=64.  Outputs: O [B,H,T,C] fp32, attn [B,H,T,T] fp32.
// Design: per-wave 16-row Q tile. Pass1: exact row max via recomputed QK^T
// (K/mask live in 192MB L2, re-read is cheap). Pass2: P=exp(s-m) -> LDS f16
// strip (64KB/wave, 256KB/WGP of the 320KB CDNA5 LDS) + row sums.
// Pass3: O = (P @ V)/l with WMMA; V tiles staged by the CDNA5 async
// global->LDS DMA (ASYNCcnt) instead of a VGPR round-trip. Finally the
// normalized attn is streamed once (537MB write = HBM roofline floor,
// ~23us @ 23.3 TB/s).
// ---------------------------------------------------------------------------

typedef _Float16 v16h __attribute__((ext_vector_type(16)));
typedef _Float16 v8h  __attribute__((ext_vector_type(8)));
typedef float    v8f  __attribute__((ext_vector_type(8)));

constexpr int   Bsz = 2, Hn = 16, T = 2048, Cd = 64;
constexpr float QSCALE  = 0.125f;    // 1/sqrt(64) folded into Q fragment
constexpr float NEG_INF = -1.0e9f;
constexpr int   WAVES = 4;
constexpr int   BLOCK = WAVES * 32;
constexpr int   QROWS = WAVES * 16;
constexpr int   VSTRIDE = Cd + 1;    // padded f32 row stride (bank rotation)

// Dynamic LDS layout
constexpr size_t LDS_P_BYTES = (size_t)WAVES * 16 * T * sizeof(_Float16); // 256KB
constexpr size_t LDS_V_OFF   = LDS_P_BYTES;
constexpr size_t LDS_V_BYTES = (size_t)32 * VSTRIDE * sizeof(float);      // 8.3KB
constexpr size_t LDS_INV_OFF = LDS_V_OFF + LDS_V_BYTES;
constexpr size_t LDS_TOTAL   = LDS_INV_OFF + (size_t)WAVES * 16 * sizeof(float);

// Compute one masked 16x16 score tile: s = (Q/8) @ K^T with mask -> -1e9.
// C/D layout: VGPR r holds (M = r + 8*half, N = n0 + (lane&15)).
__device__ __forceinline__ v8f score_tile(const float* __restrict__ kb,
                                          const int*   __restrict__ mb,
                                          const v16h*  aq,
                                          int q0, int n0, int lane) {
  const int half = lane >> 4;
  const int l16  = lane & 15;
  v8f s = {};
#pragma unroll
  for (int f = 0; f < 2; ++f) {
    // B fragment (32x16): lane column n = n0+l16; elements 0..7 -> c0+8*half+e, 8..15 -> c0+16+8*half+(e-8)
    const float* kr = kb + (size_t)(n0 + l16) * Cd + f * 32 + half * 8;
    v16h bk;
#pragma unroll
    for (int e = 0; e < 16; ++e) bk[e] = (_Float16)kr[e < 8 ? e : e + 8];
    s = __builtin_amdgcn_wmma_f32_16x16x32_f16(false, aq[f], false, bk,
                                               (short)0, s, false, false);
  }
  const int ncol = n0 + l16;
#pragma unroll
  for (int r = 0; r < 8; ++r) {
    const int row = q0 + half * 8 + r;
    const int mv  = mb[(size_t)row * T + ncol];
    s[r] = (mv == 0) ? NEG_INF : s[r];
  }
  return s;
}

__global__ void sdpa_wmma_kernel(const float* __restrict__ q,
                                 const float* __restrict__ k,
                                 const float* __restrict__ v,
                                 const int*   __restrict__ mask,
                                 float* __restrict__ out,
                                 float* __restrict__ attn) {
  extern __shared__ char smem[];
  _Float16* Pst  = (_Float16*)smem;
  float*    Vf   = (float*)(smem + LDS_V_OFF);       // 32 x VSTRIDE f32, row-major
  float*    Linv = (float*)(smem + LDS_INV_OFF);

  const int lane = threadIdx.x & 31;
  const int wave = threadIdx.x >> 5;
  const int half = lane >> 4;
  const int l16  = lane & 15;

  const int bh   = blockIdx.x / (T / QROWS);
  const int qblk = blockIdx.x % (T / QROWS);
  const int b    = bh / Hn;
  const int q0   = qblk * QROWS + wave * 16;

  const float* qb = q + (size_t)bh * T * Cd;
  const float* kb = k + (size_t)bh * T * Cd;
  const float* vb = v + (size_t)bh * T * Cd;
  const int*   mb = mask + (size_t)b * T * T;   // mask broadcast over heads

  _Float16* Pw = Pst + (size_t)wave * 16 * T;

  // ---- Q A-fragments (16x32 f16), scale folded in.
  // lane<16: M=lane, elements 0..7 -> c0+8*half+e, 8..15 -> c0+16+8*half+(e-8)
  v16h aq[2];
#pragma unroll
  for (int f = 0; f < 2; ++f) {
    const float* row = qb + (size_t)(q0 + l16) * Cd + f * 32 + half * 8;
#pragma unroll
    for (int e = 0; e < 8; ++e) {
      aq[f][e]     = (_Float16)(row[e]      * QSCALE);
      aq[f][8 + e] = (_Float16)(row[16 + e] * QSCALE);
    }
  }

  // ---- Pass 1: exact row max (per-lane partial max; one shfl tree at end)
  float mrow[8];
#pragma unroll
  for (int r = 0; r < 8; ++r) mrow[r] = -3.0e38f;
  for (int n0 = 0; n0 < T; n0 += 16) {
    __builtin_prefetch(kb + (size_t)(n0 + 16) * Cd, 0, 0);
    v8f s = score_tile(kb, mb, aq, q0, n0, lane);
#pragma unroll
    for (int r = 0; r < 8; ++r) mrow[r] = fmaxf(mrow[r], s[r]);
  }
#pragma unroll
  for (int r = 0; r < 8; ++r) {
#pragma unroll
    for (int xm = 1; xm <= 8; xm <<= 1)     // stays inside 16-lane half (wave32)
      mrow[r] = fmaxf(mrow[r], __shfl_xor(mrow[r], xm, 32));
  }

  // ---- Pass 2: P = exp(s - m) -> LDS f16 strip, accumulate row sums
  float lsum[8];
#pragma unroll
  for (int r = 0; r < 8; ++r) lsum[r] = 0.0f;
  for (int n0 = 0; n0 < T; n0 += 16) {
    v8f s = score_tile(kb, mb, aq, q0, n0, lane);
#pragma unroll
    for (int r = 0; r < 8; ++r) {
      const float p = __expf(s[r] - mrow[r]);
      lsum[r] += p;
      Pw[(size_t)(half * 8 + r) * T + (n0 + l16)] = (_Float16)p;
    }
  }
#pragma unroll
  for (int r = 0; r < 8; ++r) {
#pragma unroll
    for (int xm = 1; xm <= 8; xm <<= 1)
      lsum[r] += __shfl_xor(lsum[r], xm, 32);
  }
  if (l16 == 0) {                            // lane 0 -> rows 0..7, lane 16 -> 8..15
#pragma unroll
    for (int r = 0; r < 8; ++r)
      Linv[wave * 16 + half * 8 + r] = 1.0f / lsum[r];
  }

  // ---- Pass 3: O = (P @ V) * (1/l); V staged via async global->LDS DMA
  v8f oacc[4] = {};
  // LDS byte offset of the V tile (low 32 bits of the shared aperture address)
  const unsigned vldsbase = (unsigned)(size_t)(void*)Vf;
  for (int n0 = 0; n0 < T; n0 += 32) {
    __syncthreads();
    // Async-copy V[n0..n0+31][0..63] f32 row-major into padded LDS tile.
    // 32 rows x 64 floats = 512 x 16B chunks; 128 threads x 4 chunks each.
    {
      const float* gsrc = vb + (size_t)n0 * Cd;
#pragma unroll
      for (int i = 0; i < 4; ++i) {
        const int chunk = i * BLOCK + threadIdx.x;        // 0..511
        const int rowv  = chunk >> 4;                     // 16 chunks per row
        const int cchk  = chunk & 15;
        const unsigned ldsoff = vldsbase +
            (unsigned)(rowv * VSTRIDE * 4 + cchk * 16);
        const unsigned goff = (unsigned)chunk * 16u;      // bytes, row-major
        asm volatile("global_load_async_to_lds_b128 %0, %1, %2"
                     :: "v"(ldsoff), "v"(goff), "s"(gsrc)
                     : "memory");
      }
    }
    // A fragment from P strip (overlaps with the in-flight async copy)
    v16h ap;
    {
      const _Float16* pr = Pw + (size_t)l16 * T + n0 + half * 8;
      v8h lo = *(const v8h*)pr;
      v8h hi = *(const v8h*)(pr + 16);
      ap = __builtin_shufflevector(lo, hi, 0, 1, 2, 3, 4, 5, 6, 7,
                                   8, 9, 10, 11, 12, 13, 14, 15);
    }
    asm volatile("s_wait_asynccnt 0" ::: "memory");
    __syncthreads();
#pragma unroll
    for (int g = 0; g < 4; ++g) {
      // B fragment (32x16): lane column c = g*16+l16; element e<8 -> row 8*half+e, e>=8 -> row 16+8*half+(e-8)
      const float* vc = Vf + (size_t)(half * 8) * VSTRIDE + g * 16 + l16;
      v16h bv;
#pragma unroll
      for (int e = 0; e < 16; ++e) bv[e] = (_Float16)vc[(size_t)(e < 8 ? e : e + 8) * VSTRIDE];
      oacc[g] = __builtin_amdgcn_wmma_f32_16x16x32_f16(false, ap, false, bv,
                                                       (short)0, oacc[g], false, false);
    }
  }
  __syncthreads();

  // ---- normalize + store O
  float linv[8];
#pragma unroll
  for (int r = 0; r < 8; ++r) linv[r] = Linv[wave * 16 + half * 8 + r];
  float* ob = out + (size_t)bh * T * Cd;
#pragma unroll
  for (int g = 0; g < 4; ++g) {
#pragma unroll
    for (int r = 0; r < 8; ++r)
      ob[(size_t)(q0 + half * 8 + r) * Cd + g * 16 + l16] = oacc[g][r] * linv[r];
  }

  // ---- stream normalized attention rows (the 537MB HBM-bound write)
  float* ab = attn + (size_t)bh * T * T;
  for (int m = 0; m < 16; ++m) {
    const float inv = Linv[wave * 16 + m];
    const _Float16* pr = Pw + (size_t)m * T;
    for (int col = lane; col < T; col += 32)
      ab[(size_t)(q0 + m) * T + col] = (float)pr[col] * inv;
  }
}

extern "C" void kernel_launch(void* const* d_in, const int* in_sizes, int n_in,
                              void* d_out, int out_size, void* d_ws, size_t ws_size,
                              hipStream_t stream) {
  const float* q    = (const float*)d_in[0];
  const float* k    = (const float*)d_in[1];
  const float* v    = (const float*)d_in[2];
  const int*   mask = (const int*)d_in[3];
  float* out  = (float*)d_out;
  float* attn = out + (size_t)Bsz * Hn * T * Cd;   // outputs concatenated flat

  (void)in_sizes; (void)n_in; (void)out_size; (void)d_ws; (void)ws_size;

  hipFuncSetAttribute((const void*)sdpa_wmma_kernel,
                      hipFuncAttributeMaxDynamicSharedMemorySize, (int)LDS_TOTAL);

  dim3 grid(Bsz * Hn * (T / QROWS));
  sdpa_wmma_kernel<<<grid, BLOCK, LDS_TOTAL, stream>>>(q, k, v, mask, out, attn);
}
